// UniversalReadout_86303072845932
// MI455X (gfx1250) — matrix operand order repaired and checked by reading, not compiled, not run on hardware
//
#include <hip/hip_runtime.h>

// ---------------------------------------------------------------------------
// UniversalReadout (DeepSets readout) for MI455X / gfx1250, bf16 WMMA path.
//   phi:  x[N,256] -> 512 -> rownorm/lrelu -> 256 -> rownorm/lrelu
//   segment_sum over 128 graphs
//   rho:  [128,256] -> 512 -> rownorm/lrelu -> 128 -> rownorm/lrelu
// Native __bf16 converts (hardware v_cvt) instead of manual integer RNE.
// ---------------------------------------------------------------------------

typedef __attribute__((ext_vector_type(16))) __bf16 v16bf;
typedef __attribute__((ext_vector_type(8)))  float  v8f;

__device__ __forceinline__ float lrelu(float x) { return x > 0.0f ? x : 0.01f * x; }

__device__ __forceinline__ v8f wmma_bf16(v16bf a, v16bf b, v8f c) {
  return __builtin_amdgcn_wmma_f32_16x16x32_bf16(
      false, a, false, b, (short)0, c, false, false);
}

__device__ __forceinline__ void atomic_add_f32(float* p, float v) {
  // Guarantee the HW fp atomic (no CAS fallback).
  asm volatile("global_atomic_add_f32 %0, %1, off" :: "v"(p), "v"(v) : "memory");
}

// ---------------------------------------------------------------------------
// Weight f32 -> bf16 conversion (tiny, weights then live in L2 as bf16).
// ---------------------------------------------------------------------------
__global__ void cvt_bf16_kernel(const float* __restrict__ src,
                                __bf16* __restrict__ dst, int n) {
  int i = blockIdx.x * blockDim.x + threadIdx.x;
  if (i < n) dst[i] = (__bf16)src[i];
}

// ---------------------------------------------------------------------------
// Fused phi MLP + segment-sum. 4 waves/block, 16 rows per wave.
// ---------------------------------------------------------------------------
__global__ __launch_bounds__(128) void phi_segsum_kernel(
    const float* __restrict__ feat,   // [N,256] f32
    const __bf16* __restrict__ W1,    // [512,256] bf16
    const __bf16* __restrict__ W2,    // [256,512] bf16
    const int* __restrict__ seg,      // [N]
    float* __restrict__ sums)         // [128,256] f32 (atomic dst)
{
  constexpr int IN = 256, H1 = 512, H2 = 256;
  __shared__ __align__(16) __bf16 s_a1[4][16 * H1]; // 16KB/wave
  __shared__ float s_mean[4][16];
  __shared__ float s_rsq[4][16];

  const int wv   = threadIdx.x >> 5;
  const int lane = threadIdx.x & 31;
  const int hf   = lane >> 4;     // which 16-lane half
  const int ln   = lane & 15;
  const long r0  = (long)blockIdx.x * 64 + (long)wv * 16;

  // ---- A fragments for GEMM1 (ISA 16-bit A 16x32 layout), f32 -> bf16 ----
  v16bf af[8];
  {
    const float* fr = feat + (r0 + ln) * IN;
#pragma unroll
    for (int kk = 0; kk < 8; ++kk) {
      const int k1 = kk * 32 + hf * 8;
      float4 c0 = *(const float4*)(fr + k1);
      float4 c1 = *(const float4*)(fr + k1 + 4);
      float4 c2 = *(const float4*)(fr + k1 + 16);
      float4 c3 = *(const float4*)(fr + k1 + 20);
      v16bf a;
      a[0]=(__bf16)c0.x;  a[1]=(__bf16)c0.y;  a[2]=(__bf16)c0.z;  a[3]=(__bf16)c0.w;
      a[4]=(__bf16)c1.x;  a[5]=(__bf16)c1.y;  a[6]=(__bf16)c1.z;  a[7]=(__bf16)c1.w;
      a[8]=(__bf16)c2.x;  a[9]=(__bf16)c2.y;  a[10]=(__bf16)c2.z; a[11]=(__bf16)c2.w;
      a[12]=(__bf16)c3.x; a[13]=(__bf16)c3.y; a[14]=(__bf16)c3.z; a[15]=(__bf16)c3.w;
      af[kk] = a;
    }
  }

  float psum[8], psq[8];
#pragma unroll
  for (int g = 0; g < 8; ++g) { psum[g] = 0.f; psq[g] = 0.f; }

  // ---- GEMM1: h1[16,512] = x @ W1^T ----
  for (int nt = 0; nt < 32; ++nt) {
    const __bf16* wr = W1 + (nt * 16 + ln) * IN + hf * 8;
    v8f acc = {0.f, 0.f, 0.f, 0.f, 0.f, 0.f, 0.f, 0.f};
#pragma unroll
    for (int kk = 0; kk < 8; ++kk) {
      v16bf b;
      __builtin_memcpy(&b, wr + kk * 32, 16);
      __builtin_memcpy((char*)&b + 16, wr + kk * 32 + 16, 16);
      acc = wmma_bf16(af[kk], b, acc);
    }
    const int cb = nt * 16 + ln;
#pragma unroll
    for (int g = 0; g < 8; ++g) {
      float v = acc[g];
      psum[g] += v; psq[g] += v * v;
      s_a1[wv][(hf * 8 + g) * H1 + cb] = (__bf16)v;
    }
  }
  // column reduction across the 16 lanes of each half (wave32 shuffles)
#pragma unroll
  for (int g = 0; g < 8; ++g) {
#pragma unroll
    for (int m = 1; m < 16; m <<= 1) {
      psum[g] += __shfl_xor(psum[g], m, 32);
      psq[g]  += __shfl_xor(psq[g],  m, 32);
    }
  }
  if (ln == 0) {
#pragma unroll
    for (int g = 0; g < 8; ++g) {
      float mean = psum[g] * (1.f / H1);
      float var  = psq[g]  * (1.f / H1) - mean * mean;
      s_mean[wv][hf * 8 + g] = mean;
      s_rsq[wv][hf * 8 + g]  = __frsqrt_rn(var + 1e-5f);
    }
  }
  __syncthreads();

  // ---- rownorm + leaky-relu, in place in LDS ----
  {
    const float mean = s_mean[wv][ln];
    const float rs   = s_rsq[wv][ln];
    __bf16* rp = &s_a1[wv][ln * H1 + hf * 256];
    for (int j = 0; j < 256; ++j) {
      float v = (float)rp[j];
      rp[j] = (__bf16)lrelu((v - mean) * rs);
    }
  }
  __syncthreads();

  // ---- GEMM2: h2[16,256] = a1 @ W2^T, accumulators resident in VGPRs ----
  v8f acc2[16];
  const v8f vz = {0.f, 0.f, 0.f, 0.f, 0.f, 0.f, 0.f, 0.f};
#pragma unroll
  for (int nt = 0; nt < 16; ++nt) acc2[nt] = vz;

  for (int kk = 0; kk < 16; ++kk) {
    const __bf16* ap = &s_a1[wv][ln * H1 + kk * 32 + hf * 8];
    v16bf a;
    __builtin_memcpy(&a, ap, 16);                     // K = base .. base+7
    __builtin_memcpy((char*)&a + 16, ap + 16, 16);    // K = base+16 .. base+23
#pragma unroll
    for (int nt = 0; nt < 16; ++nt) {
      const __bf16* bp = W2 + (nt * 16 + ln) * H1 + kk * 32 + hf * 8;
      v16bf b;
      __builtin_memcpy(&b, bp, 16);
      __builtin_memcpy((char*)&b + 16, bp + 16, 16);
      acc2[nt] = wmma_bf16(a, b, acc2[nt]);
    }
  }

  float qsum[8], qsq[8];
#pragma unroll
  for (int g = 0; g < 8; ++g) { qsum[g] = 0.f; qsq[g] = 0.f; }
#pragma unroll
  for (int nt = 0; nt < 16; ++nt)
#pragma unroll
    for (int g = 0; g < 8; ++g) { float v = acc2[nt][g]; qsum[g] += v; qsq[g] += v * v; }
#pragma unroll
  for (int g = 0; g < 8; ++g) {
#pragma unroll
    for (int m = 1; m < 16; m <<= 1) {
      qsum[g] += __shfl_xor(qsum[g], m, 32);
      qsq[g]  += __shfl_xor(qsq[g],  m, 32);
    }
  }

  // ---- final rownorm + lrelu + segment atomic accumulation ----
#pragma unroll
  for (int g = 0; g < 8; ++g) {
    const int m = hf * 8 + g;
    const float mean = qsum[g] * (1.f / H2);
    const float rs   = __frsqrt_rn(qsq[g] * (1.f / H2) - mean * mean + 1e-5f);
    const int sg = seg[r0 + m] & 127;   // mask keeps atomics in-bounds
    float* dst = sums + (long)sg * H2 + ln;
#pragma unroll
    for (int nt = 0; nt < 16; ++nt) {
      float v = lrelu((acc2[nt][g] - mean) * rs);
      atomic_add_f32(dst + nt * 16, v);
    }
  }
}

// ---------------------------------------------------------------------------
// rho MLP on the 128 segment sums; one 8-wave workgroup.
// ---------------------------------------------------------------------------
__global__ __launch_bounds__(256) void rho_kernel(
    const float* __restrict__ sums,   // [128,256] f32
    const __bf16* __restrict__ W1,    // [512,256] bf16
    const __bf16* __restrict__ W2,    // [128,512] bf16
    float* __restrict__ out)          // [128,128] f32
{
  constexpr int IN = 256, H1 = 512, OUT = 128;
  __shared__ __align__(16) __bf16 s_a1[8][16 * H1];
  __shared__ float s_mean[8][16];
  __shared__ float s_rsq[8][16];

  const int wv   = threadIdx.x >> 5;
  const int lane = threadIdx.x & 31;
  const int hf   = lane >> 4;
  const int ln   = lane & 15;
  const int r0   = wv * 16;

  v16bf af[8];
  {
    const float* fr = sums + (r0 + ln) * IN;
#pragma unroll
    for (int kk = 0; kk < 8; ++kk) {
      const int k1 = kk * 32 + hf * 8;
      float4 c0 = *(const float4*)(fr + k1);
      float4 c1 = *(const float4*)(fr + k1 + 4);
      float4 c2 = *(const float4*)(fr + k1 + 16);
      float4 c3 = *(const float4*)(fr + k1 + 20);
      v16bf a;
      a[0]=(__bf16)c0.x;  a[1]=(__bf16)c0.y;  a[2]=(__bf16)c0.z;  a[3]=(__bf16)c0.w;
      a[4]=(__bf16)c1.x;  a[5]=(__bf16)c1.y;  a[6]=(__bf16)c1.z;  a[7]=(__bf16)c1.w;
      a[8]=(__bf16)c2.x;  a[9]=(__bf16)c2.y;  a[10]=(__bf16)c2.z; a[11]=(__bf16)c2.w;
      a[12]=(__bf16)c3.x; a[13]=(__bf16)c3.y; a[14]=(__bf16)c3.z; a[15]=(__bf16)c3.w;
      af[kk] = a;
    }
  }

  float psum[8], psq[8];
#pragma unroll
  for (int g = 0; g < 8; ++g) { psum[g] = 0.f; psq[g] = 0.f; }

  for (int nt = 0; nt < 32; ++nt) {
    const __bf16* wr = W1 + (nt * 16 + ln) * IN + hf * 8;
    v8f acc = {0.f, 0.f, 0.f, 0.f, 0.f, 0.f, 0.f, 0.f};
#pragma unroll
    for (int kk = 0; kk < 8; ++kk) {
      v16bf b;
      __builtin_memcpy(&b, wr + kk * 32, 16);
      __builtin_memcpy((char*)&b + 16, wr + kk * 32 + 16, 16);
      acc = wmma_bf16(af[kk], b, acc);
    }
    const int cb = nt * 16 + ln;
#pragma unroll
    for (int g = 0; g < 8; ++g) {
      float v = acc[g];
      psum[g] += v; psq[g] += v * v;
      s_a1[wv][(hf * 8 + g) * H1 + cb] = (__bf16)v;
    }
  }
#pragma unroll
  for (int g = 0; g < 8; ++g) {
#pragma unroll
    for (int m = 1; m < 16; m <<= 1) {
      psum[g] += __shfl_xor(psum[g], m, 32);
      psq[g]  += __shfl_xor(psq[g],  m, 32);
    }
  }
  if (ln == 0) {
#pragma unroll
    for (int g = 0; g < 8; ++g) {
      float mean = psum[g] * (1.f / H1);
      float var  = psq[g]  * (1.f / H1) - mean * mean;
      s_mean[wv][hf * 8 + g] = mean;
      s_rsq[wv][hf * 8 + g]  = __frsqrt_rn(var + 1e-5f);
    }
  }
  __syncthreads();

  {
    const float mean = s_mean[wv][ln];
    const float rs   = s_rsq[wv][ln];
    __bf16* rp = &s_a1[wv][ln * H1 + hf * 256];
    for (int j = 0; j < 256; ++j) {
      float v = (float)rp[j];
      rp[j] = (__bf16)lrelu((v - mean) * rs);
    }
  }
  __syncthreads();

  v8f acc2[8];
  const v8f vz = {0.f, 0.f, 0.f, 0.f, 0.f, 0.f, 0.f, 0.f};
#pragma unroll
  for (int nt = 0; nt < 8; ++nt) acc2[nt] = vz;

  for (int kk = 0; kk < 16; ++kk) {
    const __bf16* ap = &s_a1[wv][ln * H1 + kk * 32 + hf * 8];
    v16bf a;
    __builtin_memcpy(&a, ap, 16);
    __builtin_memcpy((char*)&a + 16, ap + 16, 16);
#pragma unroll
    for (int nt = 0; nt < 8; ++nt) {
      const __bf16* bp = W2 + (nt * 16 + ln) * H1 + kk * 32 + hf * 8;
      v16bf b;
      __builtin_memcpy(&b, bp, 16);
      __builtin_memcpy((char*)&b + 16, bp + 16, 16);
      acc2[nt] = wmma_bf16(a, b, acc2[nt]);
    }
  }

  float qsum[8], qsq[8];
#pragma unroll
  for (int g = 0; g < 8; ++g) { qsum[g] = 0.f; qsq[g] = 0.f; }
#pragma unroll
  for (int nt = 0; nt < 8; ++nt)
#pragma unroll
    for (int g = 0; g < 8; ++g) { float v = acc2[nt][g]; qsum[g] += v; qsq[g] += v * v; }
#pragma unroll
  for (int g = 0; g < 8; ++g) {
#pragma unroll
    for (int m = 1; m < 16; m <<= 1) {
      qsum[g] += __shfl_xor(qsum[g], m, 32);
      qsq[g]  += __shfl_xor(qsq[g],  m, 32);
    }
  }

#pragma unroll
  for (int g = 0; g < 8; ++g) {
    const int m = hf * 8 + g;
    const float mean = qsum[g] * (1.f / OUT);
    const float rs   = __frsqrt_rn(qsq[g] * (1.f / OUT) - mean * mean + 1e-5f);
#pragma unroll
    for (int nt = 0; nt < 8; ++nt) {
      float v = lrelu((acc2[nt][g] - mean) * rs);
      out[(r0 + m) * OUT + nt * 16 + ln] = v;
    }
  }
}

// ---------------------------------------------------------------------------
// Launch: prep (weights->bf16), zero sums, phi+segsum, rho.
// Workspace layout (1 MB total):
//   [0)        W1_phi bf16  262144 B
//   [262144)   W2_phi bf16  262144 B
//   [524288)   W1_rho bf16  262144 B
//   [786432)   W2_rho bf16  131072 B
//   [917504)   sums f32     131072 B
// ---------------------------------------------------------------------------
extern "C" void kernel_launch(void* const* d_in, const int* in_sizes, int n_in,
                              void* d_out, int out_size, void* d_ws, size_t ws_size,
                              hipStream_t stream) {
  (void)in_sizes; (void)n_in; (void)out_size; (void)ws_size;

  const float* feat   = (const float*)d_in[0];
  const float* W1p_f  = (const float*)d_in[1];
  const float* W2p_f  = (const float*)d_in[2];
  const float* W1r_f  = (const float*)d_in[3];
  const float* W2r_f  = (const float*)d_in[4];
  const int*   seg    = (const int*)d_in[5];
  float*       out    = (float*)d_out;

  char* ws = (char*)d_ws;
  __bf16* w1p = (__bf16*)(ws + 0);
  __bf16* w2p = (__bf16*)(ws + 262144);
  __bf16* w1r = (__bf16*)(ws + 524288);
  __bf16* w2r = (__bf16*)(ws + 786432);
  float*  sums = (float*)(ws + 917504);

  cvt_bf16_kernel<<<512, 256, 0, stream>>>(W1p_f, w1p, 512 * 256);
  cvt_bf16_kernel<<<512, 256, 0, stream>>>(W2p_f, w2p, 256 * 512);
  cvt_bf16_kernel<<<512, 256, 0, stream>>>(W1r_f, w1r, 512 * 256);
  cvt_bf16_kernel<<<256, 256, 0, stream>>>(W2r_f, w2r, 128 * 512);

  hipMemsetAsync(sums, 0, 128 * 256 * sizeof(float), stream);

  // 262144 rows / 64 rows per block
  phi_segsum_kernel<<<4096, 128, 0, stream>>>(feat, w1p, w2p, seg, sums);
  rho_kernel<<<1, 256, 0, stream>>>(sums, w1r, w2r, out);
}
